// LSTM_55164559949997
// MI455X (gfx1250) — matrix operand order repaired and checked by reading, not compiled, not run on hardware
//
#include <hip/hip_runtime.h>
#include <stdint.h>

// Problem constants (reference: T=512, B=256, DIN=256, DL=256)
#define T_STEPS 512
#define BATCH   256
#define DIN     256
#define DL      256
#define KDIM    512              // DL + DIN  (inp = concat([h, x]))
#define NDIM    1024             // 4 gates * DL
#define BT      16               // batch rows per block
#define NTHREADS 512             // 16 waves of 32
#define NPW     128              // N columns per wave (1024 / 8 N-waves)
#define NTILES  8                // 16-wide N tiles per wave
#define NHALF   4                // N tiles per register pass (pressure control)
#define KTILES  8                // 32-deep K tiles per wave (K split 2-way over waves)
#define INP_LD  (KDIM + 8)       // padded LDS row stride (bf16 elems) to dodge bank conflicts

typedef __attribute__((ext_vector_type(16))) __bf16 v16bf;
typedef __attribute__((ext_vector_type(8)))  __bf16 v8bf;
typedef __attribute__((ext_vector_type(8)))  float  v8f;

static __device__ __forceinline__ uint16_t f2bf(float f) {
  uint32_t u = __float_as_uint(f);
  u += 0x7fffu + ((u >> 16) & 1u);      // round-to-nearest-even
  return (uint16_t)(u >> 16);
}
static __device__ __forceinline__ float sigmoidf_(float x) {
  return __builtin_amdgcn_rcpf(1.0f + __expf(-x));          // v_exp + v_rcp
}
static __device__ __forceinline__ float tanhf_(float x) {
  float cx = fminf(fmaxf(x, -15.0f), 15.0f);                // avoid inf/inf
  float e  = __expf(2.0f * cx);
  return (e - 1.0f) * __builtin_amdgcn_rcpf(e + 1.0f);
}

// ---------------------------------------------------------------------------
// Prolog: pack the 4 fp32 weight matrices [256,512] into bf16, laid out in the
// per-lane WMMA B order used by this kernel (contiguous k per lane half):
//   wsB[kt][n_tile][lane][idx]  (idx = 16 contiguous bf16 per lane = 32B)
//   n = n_tile*16 + lane%16 ; k = kt*32 + (lane/16)*16 + idx
// ---------------------------------------------------------------------------
__global__ void lstm_pack_weights(const float* __restrict__ Wf,
                                  const float* __restrict__ Wi,
                                  const float* __restrict__ Wc,
                                  const float* __restrict__ Wo,
                                  uint16_t* __restrict__ wsB) {
  int id = blockIdx.x * blockDim.x + threadIdx.x;   // 0 .. 16*64*32*16-1
  int idx  =  id        & 15;
  int lane = (id >> 4)  & 31;
  int nt   = (id >> 9)  & 63;
  int kt   =  id >> 15;
  int n = nt * 16 + (lane & 15);                    // global gate-feature col
  int k = kt * 32 + ((lane >> 4) << 4) + idx;       // concat-input row
  const float* W = (n < 256) ? Wf : (n < 512) ? Wi : (n < 768) ? Wc : Wo;
  wsB[id] = f2bf(W[(size_t)(n & 255) * KDIM + k]);
}

// ---------------------------------------------------------------------------
// Persistent LSTM: 16 blocks x 512 threads. Each block owns batch rows
// [16*blockIdx.x, +16) for all 512 timesteps. 16 waves: wave = khalf*8+nwave;
// khalf selects K half (h-part vs x-part of concat), nwave selects 128 of the
// 1024 gate columns. Partial sums meet in LDS; combine phase adds bias, does
// the sigmoid/tanh/cell update and writes H,C,F,I,O.
// ---------------------------------------------------------------------------
__global__ void lstm_persistent(const float* __restrict__ X,
                                const float* __restrict__ bfp,
                                const float* __restrict__ bip,
                                const float* __restrict__ bcp,
                                const float* __restrict__ bop,
                                const uint16_t* __restrict__ wsB,
                                float* __restrict__ outH, float* __restrict__ outC,
                                float* __restrict__ outF, float* __restrict__ outI,
                                float* __restrict__ outO) {
  extern __shared__ __align__(16) uint8_t smem[];
  uint16_t (*inp)[INP_LD] = (uint16_t(*)[INP_LD])smem;                 // [BT][520] bf16 [h|x]
  float    (*cst)[DL]     = (float(*)[DL])(smem + BT*INP_LD*2);        // cell state fp32
  float    *bia           = (float*)(smem + BT*INP_LD*2 + BT*DL*4);    // [1024] gate bias
  float    (*gat)[BT][NDIM] =
      (float(*)[BT][NDIM])(smem + BT*INP_LD*2 + BT*DL*4 + NDIM*4);     // [2][16][1024]

  const int tid   = threadIdx.x;
  const int wave  = tid >> 5;
  const int lane  = tid & 31;
  const int l15   = lane & 15;
  const int lhalf = lane >> 4;          // 0 or 1 (M group / K group per ISA layout)
  const int nwave = wave & 7;           // which 128 of 1024 gate columns
  const int khalf = wave >> 3;          // which 256 of 512 K
  const int bb0   = blockIdx.x * BT;

  // bias -> LDS (used only in the combine phase)
  for (int e = tid; e < NDIM; e += NTHREADS) {
    const float* b = (e < 256) ? bfp : (e < 512) ? bip : (e < 768) ? bcp : bop;
    bia[e] = b[e & 255];
  }
  // h0 = c0 = 0; also emit H[0], C[0] = 0
  for (int e = tid; e < BT * DL; e += NTHREADS) {
    int m = e >> 8, f = e & 255;
    inp[m][f] = 0;                      // h in bf16 (low half of concat)
    cst[m][f] = 0.0f;
    size_t o0 = (size_t)(bb0 + m) * DL + f;
    outH[o0] = 0.0f;
    outC[o0] = 0.0f;
  }
  __syncthreads();

  // per-wave constant part of the packed-B byte offset: (n_tile_base*32+lane)*32
  const uint32_t bbase = ((uint32_t)(nwave * NTILES) * 32u + (uint32_t)lane) * 32u;

  union AV { v16bf v; v8bf h[2]; };

  for (int t = 0; t < T_STEPS; ++t) {
    // Defeat LICM without losing the global address space: opaque *offset*,
    // so weight loads stay global_load_b128 re-streamed from L2 each step.
    uint32_t toff = 0;
    asm volatile("" : "+s"(toff));
    const uint16_t* wsW = wsB + toff;

    // ---- phase A: stream x_t (fp32 -> bf16, non-temporal) into inp[:,256:]
    const float* xt = X + ((size_t)t * BATCH + bb0) * DIN;   // 16 contiguous rows
    for (int e = tid; e < BT * DIN; e += NTHREADS) {
      int m = e >> 8, f = e & 255;
      inp[m][DIN + f] = f2bf(__builtin_nontemporal_load(&xt[(size_t)m * DIN + f]));
    }
    // prefetch next step's x tile (16 KB contiguous, 128 cachelines)
    if (t + 1 < T_STEPS && tid < 128)
      __builtin_prefetch(X + ((size_t)(t + 1) * BATCH + bb0) * DIN + tid * 32, 0, 3);
    __syncthreads();

    // ---- phase B: per-wave [16 x 128] slice of gates = inp @ W^T ----------
    // Two register passes of 4 N-tiles each (keeps live accs at 32 VGPRs so
    // the scheduler's load lookahead doesn't spill). Within a pass: flattened
    // 32-step (kt,nt4) WMMA stream; B tiles in a 4-deep rotating buffer,
    // A tiles double-buffered from LDS.
    #pragma unroll
    for (int half = 0; half < NTILES / NHALF; ++half) {
      auto bptr = [&](int s) -> const v16bf* {
        int kt = s >> 2, nt4 = s & 3;
        int ktg = khalf * KTILES + kt;
        int n_t = half * NHALF + nt4;
        return (const v16bf*)((const uint8_t*)wsW +
                              (size_t)ktg * 65536u + (size_t)n_t * 1024u + bbase);
      };
      auto aptr = [&](int kt) -> const uint16_t* {
        int ktg = khalf * KTILES + kt;
        return &inp[l15][ktg * 32 + (lhalf << 4)];
      };

      v8f acc[NHALF];
      #pragma unroll
      for (int nt = 0; nt < NHALF; ++nt)
        #pragma unroll
        for (int r = 0; r < 8; ++r)
          acc[nt][r] = 0.0f;

      AV abuf[2];
      {
        const uint16_t* ar = aptr(0);
        abuf[0].h[0] = *(const v8bf*)(ar);
        abuf[0].h[1] = *(const v8bf*)(ar + 8);
      }
      v16bf bbuf[4];
      #pragma unroll
      for (int i = 0; i < 4; ++i) bbuf[i] = *bptr(i);

      #pragma unroll
      for (int s = 0; s < KTILES * NHALF; ++s) {
        const int kt = s >> 2, nt4 = s & 3;
        if (nt4 == 0 && kt + 1 < KTILES) {         // prefetch next A tile (LDS)
          const uint16_t* ar = aptr(kt + 1);
          abuf[(kt + 1) & 1].h[0] = *(const v8bf*)(ar);
          abuf[(kt + 1) & 1].h[1] = *(const v8bf*)(ar + 8);
        }
        acc[nt4] = __builtin_amdgcn_wmma_f32_16x16x32_bf16(
            false, abuf[kt & 1].v, false, bbuf[s & 3], (short)0, acc[nt4], false, false);
        if (s + 4 < KTILES * NHALF)                // refill slot just consumed
          bbuf[s & 3] = *bptr(s + 4);
      }

      // spill partials: C/D layout -> row m = r + 8*(lane/16), col n = lane%16
      #pragma unroll
      for (int nt = 0; nt < NHALF; ++nt) {
        int n = nwave * NPW + (half * NHALF + nt) * 16 + l15;
        #pragma unroll
        for (int r = 0; r < 8; ++r)
          gat[khalf][(lhalf << 3) + r][n] = acc[nt][r];
      }
    }
    __syncthreads();

    // ---- phase C: reduce K halves, +bias, activations, cell update, stores
    const size_t so = (size_t)t * BATCH * DL;
    for (int e = tid; e < BT * DL; e += NTHREADS) {
      int m = e >> 8, f = e & 255;
      float zf = gat[0][m][f]       + gat[1][m][f]       + bia[f];
      float zi = gat[0][m][256 + f] + gat[1][m][256 + f] + bia[256 + f];
      float zc = gat[0][m][512 + f] + gat[1][m][512 + f] + bia[512 + f];
      float zo = gat[0][m][768 + f] + gat[1][m][768 + f] + bia[768 + f];
      float fg = sigmoidf_(zf), ig = sigmoidf_(zi);
      float cc = tanhf_(zc),    og = sigmoidf_(zo);
      float cn = fg * cst[m][f] + ig * cc;
      float hn = og * tanhf_(cn);
      cst[m][f] = cn;
      inp[m][f] = f2bf(hn);                             // h for next step's A
      size_t ob = (size_t)(bb0 + m) * DL + f;
      // outputs are write-once streams: non-temporal so they don't rinse L2
      __builtin_nontemporal_store(hn, &outH[so + (size_t)BATCH * DL + ob]);
      __builtin_nontemporal_store(cn, &outC[so + (size_t)BATCH * DL + ob]);
      __builtin_nontemporal_store(fg, &outF[so + ob]);
      __builtin_nontemporal_store(ig, &outI[so + ob]);
      __builtin_nontemporal_store(og, &outO[so + ob]);
    }
    __syncthreads();
  }
}

// ---------------------------------------------------------------------------
extern "C" void kernel_launch(void* const* d_in, const int* in_sizes, int n_in,
                              void* d_out, int out_size, void* d_ws, size_t ws_size,
                              hipStream_t stream) {
  (void)in_sizes; (void)n_in; (void)out_size; (void)ws_size;
  const float* X   = (const float*)d_in[0];
  const float* Wf  = (const float*)d_in[1];
  const float* bfp = (const float*)d_in[2];
  const float* Wi  = (const float*)d_in[3];
  const float* bip = (const float*)d_in[4];
  const float* Wc  = (const float*)d_in[5];
  const float* bcp = (const float*)d_in[6];
  const float* Wo  = (const float*)d_in[7];
  const float* bop = (const float*)d_in[8];

  uint16_t* wsB = (uint16_t*)d_ws;   // 1 MB packed bf16 weights, WMMA B-layout

  float* outH = (float*)d_out;
  float* outC = outH + (size_t)(T_STEPS + 1) * BATCH * DL;
  float* outF = outC + (size_t)(T_STEPS + 1) * BATCH * DL;
  float* outI = outF + (size_t)T_STEPS * BATCH * DL;
  float* outO = outI + (size_t)T_STEPS * BATCH * DL;

  // 16 k-tiles * 64 n-tiles * 32 lanes * 16 bf16 = 524288 elements
  lstm_pack_weights<<<2048, 256, 0, stream>>>(Wf, Wi, Wc, Wo, wsB);

  size_t lds = (size_t)BT * INP_LD * 2    // inp  (bf16)     16640 B
             + (size_t)BT * DL * 4        // cell state      16384 B
             + (size_t)NDIM * 4           // bias             4096 B
             + (size_t)2 * BT * NDIM * 4; // gate partials  131072 B
  lstm_persistent<<<BATCH / BT, NTHREADS, lds, stream>>>(
      X, bfp, bip, bcp, bop, wsB, outH, outC, outF, outI, outO);
}
